// SCSSystem_78108275245660
// MI455X (gfx1250) — matrix operand order repaired and checked by reading, not compiled, not run on hardware
//
#include <hip/hip_runtime.h>

// ---------------------------------------------------------------------------
// AxonalConnections forward on MI455X (gfx1250, wave32, WMMA).
//
// For each (conn n, patch p): C[b,t] = gate[n,p] * sum_s A[b,s] * W[t,s]
//   A = patchified binary spikes  [64 x 256]  (exact in f16)
//   W = transforms[n,p]           [256 x 256] row-major (t,s), f32 in HBM
// Memory-bound: transforms = 1.07 GB streamed exactly once -> ~69 us floor
// at 23.3 TB/s including xs + out. Compute (34 GFLOP) is negligible for
// v_wmma_f32_16x16x32_f16, so the kernel is organized purely around a
// single coalesced pass over W with in-register f32->f16 conversion.
// ---------------------------------------------------------------------------

typedef __attribute__((ext_vector_type(16))) _Float16 v16h;
typedef __attribute__((ext_vector_type(8)))  _Float16 v8h;
typedef __attribute__((ext_vector_type(8)))  float    v8f;
typedef __attribute__((ext_vector_type(4)))  float    v4f;

#define N_CONNS   4
#define NPATCH    1024      // 32 x 32 patches
#define PS        256       // elems per patch (16x16)
#define BATCH     64
#define HW        512
#define A_STRIDE  264       // 256 + 8 halves pad -> conflict-free ds_load_b128

__global__ __launch_bounds__(256)
void axonal_wmma_kernel(const float* __restrict__ xs,         // [4,64,512,512]
                        const float* __restrict__ gates,      // [4,1024]
                        const float* __restrict__ transforms, // [4,1024,256,256]
                        float* __restrict__ out)              // [4,64,512,512]
{
    __shared__ _Float16 Alds[BATCH * A_STRIDE];   // ~33 KB

    const int wg   = blockIdx.x;       // 0..4095 == conn*1024 + p
    const int conn = wg >> 10;
    const int p    = wg & 1023;
    const int ph   = p >> 5;           // patch row in image
    const int pw   = p & 31;           // patch col in image

    const int tid  = threadIdx.x;
    const int wave = tid >> 5;         // 0..7
    const int lane = tid & 31;
    const int l15  = lane & 15;
    const int hi   = lane >> 4;        // half-wave select

    // ------------------------------------------------------------------
    // Stage A = patchify(xs)[conn, :, p, :] into LDS as f16 (exact: 0/1).
    // 1024 rows of 16 contiguous f32; each thread handles 4 rows.
    // ------------------------------------------------------------------
    const float* xsn = xs + (size_t)conn * BATCH * HW * HW;
    for (int q = 0; q < 4; ++q) {
        const int row_id = tid + 256 * q;        // (b, i) pair
        const int b = row_id >> 4;
        const int i = row_id & 15;
        const v4f* src = (const v4f*)(xsn + (size_t)b * HW * HW
                                          + (size_t)(ph * 16 + i) * HW
                                          + pw * 16);
        _Float16* dst = &Alds[b * A_STRIDE + i * 16];
        #pragma unroll
        for (int c = 0; c < 4; ++c) {
            v4f v = src[c];
            dst[4 * c + 0] = (_Float16)v[0];
            dst[4 * c + 1] = (_Float16)v[1];
            dst[4 * c + 2] = (_Float16)v[2];
            dst[4 * c + 3] = (_Float16)v[3];
        }
    }
    __syncthreads();

    // ------------------------------------------------------------------
    // Each wave owns N-tiles {2*wave, 2*wave+1} (== patch rows i) and all
    // 4 M-tiles (batch). W is read exactly once per workgroup.
    // ------------------------------------------------------------------
    const float* Wg = transforms + (size_t)wg * (PS * PS);   // W[t][s]

    v8f acc[4][2] = {};   // D/C f32 accumulators, 64 VGPRs

    for (int kc = 0; kc < 8; ++kc) {
        const int k0 = kc * 32;

        // A fragments (ISA 16-bit A layout: lane l15 = M row;
        // halves 0..7 -> K = k0+hi*8.., halves 8..15 -> K = k0+16+hi*8..)
        v16h a[4];
        #pragma unroll
        for (int mt = 0; mt < 4; ++mt) {
            const int row = mt * 16 + l15;
            v8h lo  = *(const v8h*)&Alds[row * A_STRIDE + k0 + hi * 8];
            v8h hi8 = *(const v8h*)&Alds[row * A_STRIDE + k0 + 16 + hi * 8];
            a[mt] = __builtin_shufflevector(lo, hi8,
                        0,1,2,3,4,5,6,7,8,9,10,11,12,13,14,15);
        }

        // B fragments (lane l15 = N column; halves 0..7 -> K = k0+hi*8..,
        // halves 8..15 -> K = k0+16+hi*8..). B[k][n] = W[n][k].
        const int kB = k0 + hi * 8;
        v16h bf[2];
        #pragma unroll
        for (int j = 0; j < 2; ++j) {
            const int n0 = (wave * 2 + j) * 16;
            const float* col = Wg + (size_t)(n0 + l15) * PS + kB;
            v16h bv;
            #pragma unroll
            for (int h = 0; h < 16; ++h)
                bv[h] = (_Float16)col[h < 8 ? h : h + 8];   // f32 -> f16 in-reg
            bf[j] = bv;
        }

        #pragma unroll
        for (int mt = 0; mt < 4; ++mt)
            #pragma unroll
            for (int j = 0; j < 2; ++j)
                acc[mt][j] = __builtin_amdgcn_wmma_f32_16x16x32_f16(
                    false, a[mt], false, bf[j],
                    (short)0, acc[mt][j], false, false);
    }

    // ------------------------------------------------------------------
    // Gate + unpatchify store. C/D layout: VGPR r -> M = r (lanes 0-15)
    // or M = 8+r (lanes 16-31); lane l15 -> N. N-tile nt == patch row i,
    // so each VGPR store writes a contiguous 64B output row segment.
    // ------------------------------------------------------------------
    const float g = gates[conn * NPATCH + p];
    float* outn = out + (size_t)conn * BATCH * HW * HW;

    #pragma unroll
    for (int mt = 0; mt < 4; ++mt) {
        #pragma unroll
        for (int j = 0; j < 2; ++j) {
            const int nt = wave * 2 + j;                 // patch row i
            float* orow = outn + (size_t)(ph * 16 + nt) * HW + pw * 16 + l15;
            #pragma unroll
            for (int r = 0; r < 8; ++r) {
                const int b = mt * 16 + hi * 8 + r;
                orow[(size_t)b * HW * HW] = acc[mt][j][r] * g;
            }
        }
    }
}

extern "C" void kernel_launch(void* const* d_in, const int* in_sizes, int n_in,
                              void* d_out, int out_size, void* d_ws, size_t ws_size,
                              hipStream_t stream) {
    (void)in_sizes; (void)n_in; (void)out_size; (void)d_ws; (void)ws_size;
    const float* xs         = (const float*)d_in[0];
    const float* gates      = (const float*)d_in[1];
    const float* transforms = (const float*)d_in[2];
    float* out              = (float*)d_out;

    dim3 grid(N_CONNS * NPATCH);   // 4096 workgroups, one per (conn, patch)
    dim3 block(256);               // 8 wave32 waves
    axonal_wmma_kernel<<<grid, block, 0, stream>>>(xs, gates, transforms, out);
}
